// ACARHead_84301618086558
// MI455X (gfx1250) — compile-verified, matrix operand order checked by reading, not run on hardware
//
#include <hip/hip_runtime.h>

// ---------------------------------------------------------------------------
// Problem constants (from reference):  N=256, C=512, T=4, H=8, W=8
//   S = T*H*W = 256 spatial locations, M = N*S = 65536 conv-GEMM rows
// Padded spatial plane: 10x10 (1-cell zero halo) -> 100 rows per (n,t).
// ---------------------------------------------------------------------------
#define NB    256       // batch
#define CC    512       // channels
#define SS    256       // T*H*W
#define MM    65536     // NB*SS
#define PROWS 102400    // NB*T*10*10 padded rows

typedef __bf16 bf16_t;
typedef __attribute__((ext_vector_type(8)))  __bf16 bf16x8;
typedef __attribute__((ext_vector_type(16))) __bf16 bf16x16;
typedef __attribute__((ext_vector_type(8)))  float  f32x8;
typedef __attribute__((ext_vector_type(4)))  int    i32x4;

union Frag16 { bf16x16 v; struct { bf16x8 lo, hi; } h; };

// A/B fragment (16x32 / 32x16 bf16) per CDNA5 ISA layout:
//   lane<16 : row = lane,    K-halves at {koff=0,  koff+16}
//   lane>=16: row = lane-16, K-halves at {koff=8,  koff+24}
// "p" points at (row, koff) in a [row][k]-contiguous layout.
__device__ __forceinline__ bf16x16 frag_load(const bf16_t* p) {
    Frag16 f;
    f.h.lo = *(const bf16x8*)(p);
    f.h.hi = *(const bf16x8*)(p + 16);
    return f.v;
}
__device__ __forceinline__ f32x8 wmma_bf16(bf16x16 a, bf16x16 b, f32x8 c) {
    return __builtin_amdgcn_wmma_f32_16x16x32_bf16(
        /*neg_a=*/false, a, /*neg_b=*/false, b,
        /*c_mod=*/(short)0, c, /*reuse_a=*/false, /*reuse_b=*/false);
}
__device__ __forceinline__ f32x8 f32x8_zero() { f32x8 z = {}; return z; }

#define BSH_STRIDE 520   // 512 + 8 bf16 pad: lane stride 1040B = 4 banks apart

// --- optional gfx1250 async global->LDS path (ASYNCcnt), guarded -----------
#if defined(__gfx1250__) && __has_builtin(__builtin_amdgcn_global_load_async_to_lds_b128)
#define HAVE_ASYNC_LDS 1
#else
#define HAVE_ASYNC_LDS 0
#endif

#if HAVE_ASYNC_LDS
typedef __attribute__((address_space(1))) i32x4* as1_i32x4p;   // 64-bit global
typedef __attribute__((address_space(3))) i32x4* as3_i32x4p;   // 32-bit LDS

__device__ __forceinline__ void async_copy_b128(const void* g, void* l) {
    // AS1 pointer via 64-bit integer round-trip (value-preserving);
    // AS3 pointer = low 32 bits of the generic LDS address (ISA 10.2).
    __builtin_amdgcn_global_load_async_to_lds_b128(
        (as1_i32x4p)(unsigned long long)g,
        (as3_i32x4p)(unsigned int)(unsigned long long)l,
        0, 0);
}
__device__ __forceinline__ void async_wait0() {
#if __has_builtin(__builtin_amdgcn_s_wait_asynccnt)
    __builtin_amdgcn_s_wait_asynccnt(0);
#else
    asm volatile("s_wait_asynccnt 0x0" ::: "memory");
#endif
}
#endif

// ---------------------------------------------------------------------------
// P1: x [n][c][s] f32 -> xp [(n*4+t)*100 + (h+1)*10 + (w+1)][c] bf16, halo=0
// ---------------------------------------------------------------------------
__global__ __launch_bounds__(256)
void k_make_xt_pad(const float* __restrict__ x, bf16_t* __restrict__ xp) {
    const size_t total  = (size_t)PROWS * CC;
    const size_t stride = (size_t)gridDim.x * blockDim.x;
    for (size_t idx = (size_t)blockIdx.x * blockDim.x + threadIdx.x;
         idx < total; idx += stride) {
        const int c   = (int)(idx & 511);
        const int row = (int)(idx >> 9);
        const int pr  = row % 100;
        const int nt  = row / 100;
        const int hh  = pr / 10, ww = pr % 10;
        const int n   = nt >> 2, t = nt & 3;
        float v = 0.f;
        if (hh >= 1 && hh <= 8 && ww >= 1 && ww <= 8) {
            const int s = (t << 6) + ((hh - 1) << 3) + (ww - 1);
            v = x[((size_t)n * CC + c) * SS + s];
        }
        xp[idx] = (bf16_t)v;
    }
}

// ---------------------------------------------------------------------------
// P2: W [co][ci][9] f32  ->  wt [tap][co][ci] bf16  (B fragments are [n][k])
// ---------------------------------------------------------------------------
__global__ __launch_bounds__(256)
void k_make_wt(const float* __restrict__ w, bf16_t* __restrict__ wt) {
    const size_t total  = (size_t)9 * CC * CC;
    const size_t stride = (size_t)gridDim.x * blockDim.x;
    for (size_t idx = (size_t)blockIdx.x * blockDim.x + threadIdx.x;
         idx < total; idx += stride) {
        const int ci = (int)(idx & 511);
        const int co = (int)((idx >> 9) & 511);
        const int r  = (int)(idx >> 18);
        wt[idx] = (bf16_t)w[((size_t)co * CC + ci) * 9 + r];
    }
}

// ---------------------------------------------------------------------------
// Conv core: 9 accumulated GEMMs (K=512/tap), B slab staged in LDS per tap
// (async global->LDS when available). Block: 8 waves, tile 128(M) x 64(Cout).
// EPI 0 -> out bf16 [s][n][co]    (q, k)
// EPI 1 -> out bf16 [s][co][n]    (v transposed)
// EPI 2 -> out f32  [n][co][s] = xres + acc  (final conv + residual)
// ---------------------------------------------------------------------------
template <int EPI>
__device__ __forceinline__
void conv_body(const bf16_t* __restrict__ xp, const bf16_t* __restrict__ wt,
               bf16_t* __restrict__ outb, const float* __restrict__ xres,
               float* __restrict__ outf) {
    __shared__ __align__(16) bf16_t bsh[64 * BSH_STRIDE];

    const int tid   = threadIdx.x;
    const int lane  = tid & 31;
    const int wave  = tid >> 5;
    const int m0    = blockIdx.x * 128 + wave * 16;
    const int n0    = blockIdx.y * 64;
    const int row   = lane & 15;
    const int koff  = (lane >> 4) << 3;
    const int col   = lane & 15;
    const int rbase = (lane >> 4) << 3;

    const int nIdx  = m0 >> 8;      // batch sample (constant within tile)
    const int sBase = m0 & 255;
    const int s = sBase + row;
    const int t = s >> 6, h = (s >> 3) & 7, w = s & 7;
    const int ntBase = (nIdx * 4 + t) * 100;

    f32x8 acc[4];
    #pragma unroll
    for (int i = 0; i < 4; ++i) acc[i] = f32x8_zero();

    for (int r = 0; r < 9; ++r) {
        // ---- stage B slab [64 co][512 ci] -> LDS (row stride 520) ----
        __syncthreads();                       // previous tap's reads done
        #pragma unroll
        for (int chunk = 0; chunk < 16; ++chunk) {
            const int flat = chunk * 2048 + tid * 8;
            const int co = flat >> 9, ci = flat & 511;
            const bf16_t* src = wt + (((size_t)r << 9) + n0 + co) * CC + ci;
            bf16_t*       dst = &bsh[co * BSH_STRIDE + ci];
#if HAVE_ASYNC_LDS
            async_copy_b128(src, dst);          // global -> LDS, no VGPRs
#else
            *(bf16x8*)dst = *(const bf16x8*)src;
#endif
        }
#if HAVE_ASYNC_LDS
        async_wait0();
#endif
        __syncthreads();

        const int dh = r / 3 - 1, dw = r % 3 - 1;
        const bf16_t* arow = xp +
            ((size_t)(ntBase + (h + 1 + dh) * 10 + (w + 1 + dw)) << 9) + koff;
        if (r < 8) {    // hint next tap's A row into cache (global_prefetch)
            const int r2 = r + 1;
            const int dh2 = r2 / 3 - 1, dw2 = r2 % 3 - 1;
            __builtin_prefetch(
                xp + ((size_t)(ntBase + (h + 1 + dh2) * 10 + (w + 1 + dw2)) << 9)
                   + koff, 0, 1);
        }
        #pragma unroll 4
        for (int k0 = 0; k0 < CC; k0 += 32) {
            const bf16x16 a = frag_load(arow + k0);
            #pragma unroll
            for (int nt = 0; nt < 4; ++nt) {
                const bf16x16 b =
                    frag_load(&bsh[(nt * 16 + col) * BSH_STRIDE + k0 + koff]);
                acc[nt] = wmma_bf16(a, b, acc[nt]);
            }
        }
    }

    #pragma unroll
    for (int nt = 0; nt < 4; ++nt) {
        #pragma unroll
        for (int rr = 0; rr < 8; ++rr) {
            const int so = sBase + rbase + rr;
            const int co = n0 + nt * 16 + col;
            const float vv = acc[nt][rr];
            if (EPI == 0) {
                outb[((size_t)so * NB + nIdx) * CC + co] = (bf16_t)vv;
            } else if (EPI == 1) {
                outb[((size_t)so * CC + co) * NB + nIdx] = (bf16_t)vv;
            } else {
                const size_t addr = ((size_t)nIdx * CC + co) * SS + so;
                outf[addr] = xres[addr] + vv;
            }
        }
    }
}

__global__ __launch_bounds__(256, 2)
void k_conv_qk(const bf16_t* __restrict__ xp, const bf16_t* __restrict__ wt,
               bf16_t* __restrict__ out) {
    conv_body<0>(xp, wt, out, nullptr, nullptr);
}
__global__ __launch_bounds__(256, 2)
void k_conv_vt(const bf16_t* __restrict__ xp, const bf16_t* __restrict__ wt,
               bf16_t* __restrict__ out) {
    conv_body<1>(xp, wt, out, nullptr, nullptr);
}
__global__ __launch_bounds__(256, 2)
void k_conv_final(const bf16_t* __restrict__ vp, const bf16_t* __restrict__ wt,
                  const float* __restrict__ xres, float* __restrict__ out) {
    conv_body<2>(vp, wt, nullptr, xres, out);
}

// ---------------------------------------------------------------------------
// C1: att[s][i][j] = (q_s . k_s) * scale, masked by roi groups.
// Grid (256 s, 2 iChunk, 2 jChunk); block 8 waves; wave = 16 i x 128 j.
// ---------------------------------------------------------------------------
__global__ __launch_bounds__(256, 2)
void k_att(const bf16_t* __restrict__ q, const bf16_t* __restrict__ k,
           const int* __restrict__ roi, float* __restrict__ attf) {
    const int lane  = threadIdx.x & 31;
    const int wave  = threadIdx.x >> 5;
    const int sIdx  = blockIdx.x;
    const int i0    = blockIdx.y * 128 + wave * 16;
    const int j0    = blockIdx.z * 128;
    const int row   = lane & 15;
    const int koff  = (lane >> 4) << 3;
    const int col   = lane & 15;
    const int rbase = (lane >> 4) << 3;

    const bf16_t* abase = q + ((size_t)sIdx * NB + i0 + row) * CC + koff;
    const bf16_t* bbase = k + ((size_t)sIdx * NB + j0 + col) * CC + koff;

    f32x8 acc[8];
    #pragma unroll
    for (int i = 0; i < 8; ++i) acc[i] = f32x8_zero();

    #pragma unroll 2
    for (int k0 = 0; k0 < CC; k0 += 32) {
        const bf16x16 a = frag_load(abase + k0);
        #pragma unroll
        for (int jt = 0; jt < 8; ++jt) {
            const bf16x16 b = frag_load(bbase + (size_t)jt * 16 * CC + k0);
            acc[jt] = wmma_bf16(a, b, acc[jt]);
        }
    }
    const float scale = 0.044194173824159216f;  // 1/sqrt(512)
    #pragma unroll
    for (int jt = 0; jt < 8; ++jt) {
        const int jj  = j0 + jt * 16 + col;
        const int rjj = roi[jj];
        #pragma unroll
        for (int rr = 0; rr < 8; ++rr) {
            const int ii = i0 + rbase + rr;
            const float v = (roi[ii] == rjj) ? acc[jt][rr] * scale : -1e30f;
            attf[((size_t)sIdx * NB + ii) * NB + jj] = v;
        }
    }
}

// ---------------------------------------------------------------------------
// C2: row softmax over j (256 wide). One wave (32 lanes) per row, 8 vals/lane.
// ---------------------------------------------------------------------------
__global__ __launch_bounds__(256)
void k_softmax(const float* __restrict__ attf, bf16_t* __restrict__ attb) {
    const int lane = threadIdx.x & 31;
    const size_t rowId = (size_t)blockIdx.x * 8 + (threadIdx.x >> 5);
    const float* rp = attf + rowId * NB;
    float v[8];
    float m = -3.4e38f;
    #pragma unroll
    for (int e = 0; e < 8; ++e) { v[e] = rp[lane + 32 * e]; m = fmaxf(m, v[e]); }
    #pragma unroll
    for (int off = 16; off >= 1; off >>= 1) m = fmaxf(m, __shfl_xor(m, off, 32));
    float sum = 0.f;
    #pragma unroll
    for (int e = 0; e < 8; ++e) { v[e] = __expf(v[e] - m); sum += v[e]; }
    #pragma unroll
    for (int off = 16; off >= 1; off >>= 1) sum += __shfl_xor(sum, off, 32);
    const float inv = 1.0f / sum;
    bf16_t* wp = attb + rowId * NB;
    #pragma unroll
    for (int e = 0; e < 8; ++e) wp[lane + 32 * e] = (bf16_t)(v[e] * inv);
}

// ---------------------------------------------------------------------------
// C3: virt[i][c] = att[i][j] * v[j][c] per location s (K=256 over j).
// Grid (256 s, 2 iChunk, 8 cChunk); wave = 16 i x 64 c. Output [(i*256+s)][c].
// ---------------------------------------------------------------------------
__global__ __launch_bounds__(256, 2)
void k_virt(const bf16_t* __restrict__ attb, const bf16_t* __restrict__ vt,
            float* __restrict__ virt) {
    const int lane  = threadIdx.x & 31;
    const int wave  = threadIdx.x >> 5;
    const int sIdx  = blockIdx.x;
    const int i0    = blockIdx.y * 128 + wave * 16;
    const int c0    = blockIdx.z * 64;
    const int row   = lane & 15;
    const int koff  = (lane >> 4) << 3;
    const int col   = lane & 15;
    const int rbase = (lane >> 4) << 3;

    const bf16_t* abase = attb + ((size_t)sIdx * NB + i0 + row) * NB + koff;
    const bf16_t* bbase = vt   + ((size_t)sIdx * CC + c0 + col) * NB + koff;

    f32x8 acc[4];
    #pragma unroll
    for (int i = 0; i < 4; ++i) acc[i] = f32x8_zero();

    #pragma unroll 2
    for (int k0 = 0; k0 < NB; k0 += 32) {
        const bf16x16 a = frag_load(abase + k0);
        #pragma unroll
        for (int ct = 0; ct < 4; ++ct) {
            const bf16x16 b = frag_load(bbase + (size_t)ct * 16 * NB + k0);
            acc[ct] = wmma_bf16(a, b, acc[ct]);
        }
    }
    #pragma unroll
    for (int ct = 0; ct < 4; ++ct) {
        #pragma unroll
        for (int rr = 0; rr < 8; ++rr) {
            const int ii = i0 + rbase + rr;
            const int cc = c0 + ct * 16 + col;
            virt[((size_t)ii * SS + sIdx) * CC + cc] = acc[ct][rr];
        }
    }
}

// ---------------------------------------------------------------------------
// D1: per-sample GroupNorm stats over C*S = 131072 elements. LDS reduction.
// ---------------------------------------------------------------------------
__global__ __launch_bounds__(256)
void k_gnstats(const float* __restrict__ virt, float* __restrict__ stats) {
    const int n = blockIdx.x;
    const float* base = virt + (size_t)n * (CC * SS);
    float s = 0.f, ss = 0.f;
    for (int i = threadIdx.x; i < CC * SS; i += 256) {
        const float v = base[i];
        s += v; ss += v * v;
    }
    __shared__ float sh0[256];
    __shared__ float sh1[256];
    sh0[threadIdx.x] = s; sh1[threadIdx.x] = ss;
    __syncthreads();
    for (int off = 128; off > 0; off >>= 1) {
        if ((int)threadIdx.x < off) {
            sh0[threadIdx.x] += sh0[threadIdx.x + off];
            sh1[threadIdx.x] += sh1[threadIdx.x + off];
        }
        __syncthreads();
    }
    if (threadIdx.x == 0) {
        const float inv  = 1.0f / (float)(CC * SS);
        const float mean = sh0[0] * inv;
        const float var  = sh1[0] * inv - mean * mean;
        stats[2 * n]     = mean;
        stats[2 * n + 1] = rsqrtf(var + 1e-5f);
    }
}

// ---------------------------------------------------------------------------
// D2: normalize + affine + ReLU, cast bf16, write PADDED layout (halo = 0)
// so the final conv runs the same unconditional-load path.
// ---------------------------------------------------------------------------
__global__ __launch_bounds__(256)
void k_gnapply_pad(const float* __restrict__ virt, const float* __restrict__ stats,
                   const float* __restrict__ gamma, const float* __restrict__ beta,
                   bf16_t* __restrict__ vp) {
    const size_t total  = (size_t)PROWS * CC;
    const size_t stride = (size_t)gridDim.x * blockDim.x;
    for (size_t idx = (size_t)blockIdx.x * blockDim.x + threadIdx.x;
         idx < total; idx += stride) {
        const int c   = (int)(idx & 511);
        const int row = (int)(idx >> 9);
        const int pr  = row % 100;
        const int nt  = row / 100;
        const int hh  = pr / 10, ww = pr % 10;
        const int n   = nt >> 2, t = nt & 3;
        float val = 0.f;
        if (hh >= 1 && hh <= 8 && ww >= 1 && ww <= 8) {
            const int s = (t << 6) + ((hh - 1) << 3) + (ww - 1);
            const size_t vidx = ((size_t)n * SS + s) * CC + c;
            val = (virt[vidx] - stats[2 * n]) * stats[2 * n + 1];
            val = fmaxf(val * gamma[c] + beta[c], 0.0f);
        }
        vp[idx] = (bf16_t)val;
    }
}

// ---------------------------------------------------------------------------
// Host-side launch sequence (all on `stream`, graph-capture safe).
// ---------------------------------------------------------------------------
extern "C" void kernel_launch(void* const* d_in, const int* in_sizes, int n_in,
                              void* d_out, int out_size, void* d_ws, size_t ws_size,
                              hipStream_t stream) {
    const float* x     = (const float*)d_in[0];
    const int*   roi   = (const int*)  d_in[1];
    const float* Wq    = (const float*)d_in[2];
    const float* Wk    = (const float*)d_in[3];
    const float* Wv    = (const float*)d_in[4];
    const float* Wc    = (const float*)d_in[5];
    const float* gamma = (const float*)d_in[6];
    const float* beta  = (const float*)d_in[7];
    float*       out   = (float*)d_out;

    char* ws = (char*)d_ws;
    const size_t szXP  = (size_t)PROWS * CC * sizeof(bf16_t);   // 104,857,600
    const size_t szWT  = (size_t)9 * CC * CC * sizeof(bf16_t);  //   4,718,592
    const size_t szQK  = (size_t)SS * NB * CC * sizeof(bf16_t); //  67,108,864
    const size_t szATF = (size_t)SS * NB * NB * sizeof(float);  //  67,108,864
    const size_t szATB = (size_t)SS * NB * NB * sizeof(bf16_t); //  33,554,432
    const size_t szVRT = (size_t)MM * CC * sizeof(float);       // 134,217,728

    size_t off = 0;
    bf16_t* xp   = (bf16_t*)(ws + off); off += szXP;
    bf16_t* vp   = (bf16_t*)(ws + off); off += szXP;
    bf16_t* wtq  = (bf16_t*)(ws + off); off += szWT;
    bf16_t* wtk  = (bf16_t*)(ws + off); off += szWT;
    bf16_t* wtv  = (bf16_t*)(ws + off); off += szWT;
    bf16_t* wtc  = (bf16_t*)(ws + off); off += szWT;
    bf16_t* qb   = (bf16_t*)(ws + off); off += szQK;
    bf16_t* kb   = (bf16_t*)(ws + off); off += szQK;
    bf16_t* vt   = (bf16_t*)(ws + off); off += szQK;
    float*  attf = (float*) (ws + off); off += szATF;
    bf16_t* attb = (bf16_t*)(ws + off); off += szATB;
    float*  virt = (float*) (ws + off); off += szVRT;
    float*  stats= (float*) (ws + off); off += 2048;

    // P1/P2: layout transforms + bf16 conversion (+halo zero-fill)
    k_make_xt_pad<<<25600, 256, 0, stream>>>(x, xp);
    k_make_wt    <<<9216,  256, 0, stream>>>(Wq, wtq);
    k_make_wt    <<<9216,  256, 0, stream>>>(Wk, wtk);
    k_make_wt    <<<9216,  256, 0, stream>>>(Wv, wtv);
    k_make_wt    <<<9216,  256, 0, stream>>>(Wc, wtc);

    // q, k, v convs (v written transposed per location)
    dim3 gConv(MM / 128, CC / 64, 1);
    k_conv_qk<<<gConv, 256, 0, stream>>>(xp, wtq, qb);
    k_conv_qk<<<gConv, 256, 0, stream>>>(xp, wtk, kb);
    k_conv_vt<<<gConv, 256, 0, stream>>>(xp, wtv, vt);

    // attention scores + mask, softmax, att @ v
    k_att    <<<dim3(SS, 2, 2), 256, 0, stream>>>(qb, kb, roi, attf);
    k_softmax<<<8192,          256, 0, stream>>>(attf, attb);
    k_virt   <<<dim3(SS, 2, 8), 256, 0, stream>>>(attb, vt, virt);

    // GroupNorm(1 group) + ReLU -> padded bf16 activations
    k_gnstats    <<<NB,    256, 0, stream>>>(virt, stats);
    k_gnapply_pad<<<25600, 256, 0, stream>>>(virt, stats, gamma, beta, vp);

    // final conv + residual
    k_conv_final<<<gConv, 256, 0, stream>>>(vp, wtc, x, out);
}
